// ImuModel_25220047962187
// MI455X (gfx1250) — matrix operand order repaired and checked by reading, not compiled, not run on hardware
//
#include <hip/hip_runtime.h>

// ---------------------------------------------------------------------------
// Types for CDNA5 WMMA (gfx1250, wave32)
// ---------------------------------------------------------------------------
typedef __attribute__((ext_vector_type(16))) __bf16 v16bf;
typedef __attribute__((ext_vector_type(8)))  float  v8f;

union Frag16 { v16bf v; uint4 q[2]; };

__device__ __forceinline__ unsigned short f2bf(float f) {
  union { float f; unsigned int u; } x; x.f = f;
  unsigned int u = x.u;
  u += 0x7fffu + ((u >> 16) & 1u);        // round-to-nearest-even
  return (unsigned short)(u >> 16);
}
__device__ __forceinline__ float bf2f(unsigned short s) {
  union { unsigned int u; float f; } x; x.u = ((unsigned int)s) << 16;
  return x.f;
}
__device__ __forceinline__ float hsig(float x) {
  return fminf(fmaxf(0.2f * x + 0.5f, 0.0f), 1.0f);
}
// conv tap index (0..8) -> pixel offset in an 18-wide padded grid
__device__ __host__ constexpr int tap_off(int pos) { return (pos / 3) * 18 + (pos % 3); }

// Problem constants: B=16, T=64, H=W=16, C=6, F=96, gates=384, M=B*H*W=4096.
// Padded activation pixel grids are [18][18] (1-pixel zero halo).
//
// ---------------------------------------------------------------------------
// Implicit-GEMM ConvLSTM gate kernel:  g[M=4096, N=384] = im2col(A) @ Wt + bias
//   K layout: [0, KXPAD) input-conv part over padded bf16 input (chan stride
//   XCS, 9 taps), [KXPAD, KTOT) recurrent part over padded bf16 h_prev.
//   NO LDS, NO barriers: each lane loads its A-fragment directly from the
//   halo-padded buffers as aligned b128s (8-aligned K-groups never straddle a
//   tap).  K loop is fully unrolled -> every address is base + immediate.
// Block: 256 threads (8 wave32), 64x128 output tile; each wave owns a 32x32
// tile: 2 A-frags x 2 B-frags -> 4 WMMA accumulators (max fragment reuse).
// ---------------------------------------------------------------------------
template<int XCS, int KXPAD, int KTOT>
__global__ __launch_bounds__(256) void
lstm_gate_gemm(const unsigned short* __restrict__ xin,   // padded input, this t
               int xbstride,                             // b-stride of xin (elems)
               const unsigned short* __restrict__ hpad,  // [16][18][18][96] bf16
               const unsigned short* __restrict__ wt,    // [384][KTOT] bf16
               const float* __restrict__ bias,           // [384]
               float* __restrict__ g)                    // [4096][384]
{
  const int tid   = threadIdx.x;
  const int mbase = blockIdx.x * 64;
  const int nbase = blockIdx.y * 128;

  const int lane = tid & 31;
  const int half = lane >> 4;         // K-half selector per ISA layout
  const int m    = lane & 15;
  const int wave = tid >> 5;
  const int wm   = wave & 1;          // 32-row slab
  const int wn   = wave >> 1;         // 32-col slab

  // A-fragment rows for this lane (two 16-row sub-slabs)
  const int r0 = mbase + wm * 32 + m;
  const int r1 = r0 + 16;
  const int b0i = r0 >> 8, rem0 = r0 & 255, y0 = rem0 >> 4, x0 = rem0 & 15;
  const int b1i = r1 >> 8, rem1 = r1 & 255, y1 = rem1 >> 4, x1 = rem1 & 15;

  // per-lane pixel base pointers (top-left of the 3x3 halo window)
  const unsigned short* xpix0 = xin + (size_t)b0i * xbstride + (y0 * 18 + x0) * XCS;
  const unsigned short* xpix1 = xin + (size_t)b1i * xbstride + (y1 * 18 + x1) * XCS;
  const unsigned short* hpix0 = hpad + ((size_t)b0i * 324 + y0 * 18 + x0) * 96 + half * 8;
  const unsigned short* hpix1 = hpad + ((size_t)b1i * 324 + y1 * 18 + x1) * 96 + half * 8;
  const unsigned short* xph0  = xpix0 + half * 8;   // used when XCS==96
  const unsigned short* xph1  = xpix1 + half * 8;

  // B fragment base: col = lane%16, K offset = (lane/16)*8 (+16 for upper 8)
  const unsigned short* wtp0 = wt + (size_t)(nbase + wn * 32 + m) * KTOT + half * 8;
  const unsigned short* wtp1 = wtp0 + (size_t)16 * KTOT;

  v8f acc00 = {}, acc01 = {}, acc10 = {}, acc11 = {};

  // XCS==8 x-part: the tap depends on `half`; both candidates are immediates.
  auto ldx8 = [&](const unsigned short* pix, int posc) -> uint4 {
    const int  o0 = tap_off(posc) * 8;
    const int  o1 = tap_off(posc + 1) * 8;
    const int  off = half ? o1 : o0;
    const bool ok  = half ? (posc + 1 < 9) : (posc < 9);
    uint4 v = {0u, 0u, 0u, 0u};
    if (ok) v = *(const uint4*)(pix + off);
    return v;
  };

  #pragma unroll
  for (int kc = 0; kc < KTOT / 32; ++kc) {
    const int k0 = kc * 32;
    Frag16 a0, a1, b0, b1;
    if (k0 < KXPAD) {
      if (XCS == 96) {                               // layer-1 input conv
        const int pos  = k0 / 96;
        const int base = tap_off(pos) * 96 + (k0 - pos * 96);
        a0.q[0] = *(const uint4*)(xph0 + base);
        a0.q[1] = *(const uint4*)(xph0 + base + 16);
        a1.q[0] = *(const uint4*)(xph1 + base);
        a1.q[1] = *(const uint4*)(xph1 + base + 16);
      } else {                                       // layer-0 input conv (C pad 8)
        const int posL = k0 >> 3;
        a0.q[0] = ldx8(xpix0, posL);
        a0.q[1] = ldx8(xpix0, posL + 2);
        a1.q[0] = ldx8(xpix1, posL);
        a1.q[1] = ldx8(xpix1, posL + 2);
      }
    } else {                                         // recurrent conv over h_prev
      const int kk   = k0 - KXPAD;
      const int pos  = kk / 96;
      const int base = tap_off(pos) * 96 + (kk - pos * 96);
      a0.q[0] = *(const uint4*)(hpix0 + base);
      a0.q[1] = *(const uint4*)(hpix0 + base + 16);
      a1.q[0] = *(const uint4*)(hpix1 + base);
      a1.q[1] = *(const uint4*)(hpix1 + base + 16);
    }
    b0.q[0] = *(const uint4*)(wtp0 + k0);
    b0.q[1] = *(const uint4*)(wtp0 + k0 + 16);
    b1.q[0] = *(const uint4*)(wtp1 + k0);
    b1.q[1] = *(const uint4*)(wtp1 + k0 + 16);

    acc00 = __builtin_amdgcn_wmma_f32_16x16x32_bf16(false, a0.v, false, b0.v, (short)0, acc00, false, false);
    acc01 = __builtin_amdgcn_wmma_f32_16x16x32_bf16(false, a0.v, false, b1.v, (short)0, acc01, false, false);
    acc10 = __builtin_amdgcn_wmma_f32_16x16x32_bf16(false, a1.v, false, b0.v, (short)0, acc10, false, false);
    acc11 = __builtin_amdgcn_wmma_f32_16x16x32_bf16(false, a1.v, false, b1.v, (short)0, acc11, false, false);
  }

  // C/D layout: VGPR v -> row v + 8*half, col = lane%16
  const int   col0 = nbase + wn * 32 + m;
  const float bi0  = bias[col0];
  const float bi1  = bias[col0 + 16];
  #pragma unroll
  for (int v = 0; v < 8; ++v) {
    const int ra = mbase + wm * 32 + v + half * 8;
    const int rb = ra + 16;
    g[ra * 384 + col0]      = acc00[v] + bi0;
    g[ra * 384 + col0 + 16] = acc01[v] + bi1;
    g[rb * 384 + col0]      = acc10[v] + bi0;
    g[rb * 384 + col0 + 16] = acc11[v] + bi1;
  }
}

// ---------------------------------------------------------------------------
// Gate nonlinearities + LSTM cell update.  h state goes to a padded halo
// buffer (recurrent conv input); BN(h) goes either to a padded buffer (next
// layer's conv input) or flat (dense input), selected at compile time.
// ---------------------------------------------------------------------------
template<bool PADHBN>
__global__ __launch_bounds__(256) void
lstm_pointwise(const float* __restrict__ g,
               float* __restrict__ cst,            // fp32 cell state [4096*96]
               unsigned short* __restrict__ hpad,  // bf16 [16][18][18][96]
               unsigned short* __restrict__ hbn,   // bf16 padded or flat
               const float* __restrict__ scale,
               const float* __restrict__ shift)
{
  const int e = blockIdx.x * blockDim.x + threadIdx.x;
  if (e >= 4096 * 96) return;
  const int row = e / 96;
  const int f   = e - row * 96;
  const int b   = row >> 8;
  const int rem = row & 255;
  const int y   = rem >> 4;
  const int xx  = rem & 15;
  const int pidx = ((b * 324) + (y + 1) * 18 + (xx + 1)) * 96 + f;

  const float* gr = g + (size_t)row * 384;
  const float gi = hsig(gr[f]);
  const float gf = hsig(gr[96 + f]);
  const float gc = tanhf(gr[192 + f]);
  const float go = hsig(gr[288 + f]);
  const float c  = gf * cst[e] + gi * gc;
  const float h  = go * tanhf(c);
  cst[e] = c;
  hpad[pidx] = f2bf(h);
  hbn[PADHBN ? pidx : e] = f2bf(h * scale[f] + shift[f]);
}

// ---------------------------------------------------------------------------
// Per-timestep dense partial: out[b][nc] += sum_e h1bn[b][e] * Wd[t*24576+e][nc]
// ---------------------------------------------------------------------------
__global__ __launch_bounds__(256) void
dense_acc(const unsigned short* __restrict__ h1bn,  // [16][24576] bf16 (flat)
          const float* __restrict__ Wd, int t,
          float* __restrict__ out)
{
  __shared__ float red[10][257];
  const int b = blockIdx.x;
  float acc[10];
  #pragma unroll
  for (int j = 0; j < 10; ++j) acc[j] = 0.f;
  const unsigned short* hp = h1bn + b * 24576;
  const float* wp = Wd + (size_t)t * 24576 * 10;
  for (int e = threadIdx.x; e < 24576; e += 256) {
    const float hv = bf2f(hp[e]);
    const float* w = wp + (size_t)e * 10;
    #pragma unroll
    for (int j = 0; j < 10; ++j) acc[j] += hv * w[j];
  }
  #pragma unroll
  for (int j = 0; j < 10; ++j) red[j][threadIdx.x] = acc[j];
  __syncthreads();
  if (threadIdx.x < 10) {
    float s = 0.f;
    for (int i = 0; i < 256; ++i) s += red[threadIdx.x][i];
    out[b * 10 + threadIdx.x] += s;
  }
}

// ---------------------------------------------------------------------------
// Prep kernels
// ---------------------------------------------------------------------------
__global__ void zero_ws(uint4* __restrict__ p, int n) {
  const int i = blockIdx.x * blockDim.x + threadIdx.x;
  if (i < n) { uint4 z = {0u, 0u, 0u, 0u}; p[i] = z; }
}

__global__ void prep_params(const float* g0, const float* be0, const float* m0, const float* v0,
                            const float* g1, const float* be1, const float* m1, const float* v1,
                            const float* bd,
                            float* scale0, float* shift0, float* scale1, float* shift1,
                            float* out)
{
  const int i = threadIdx.x;
  if (i < 96) {
    const float s0 = g0[i] * rsqrtf(v0[i] + 1e-3f);
    scale0[i] = s0; shift0[i] = be0[i] - m0[i] * s0;
    const float s1 = g1[i] * rsqrtf(v1[i] + 1e-3f);
    scale1[i] = s1; shift1[i] = be1[i] - m1[i] * s1;
  }
  if (i < 160) out[i] = bd[i % 10];
}

// Stage x once into padded bf16 [B][T][18][18][8] (halo + chans 6,7 zero).
__global__ void pad_x(const float* __restrict__ x, unsigned short* __restrict__ xpad) {
  const int p = blockIdx.x * blockDim.x + threadIdx.x;   // pixel over B*T*H*W
  if (p >= 262144) return;
  const int bt  = p >> 8;
  const int rem = p & 255;
  const int y   = rem >> 4;
  const int xx  = rem & 15;
  const float* src = x + (size_t)p * 6;
  unsigned short* dst = xpad + ((size_t)bt * 324 + (y + 1) * 18 + (xx + 1)) * 8;
  #pragma unroll
  for (int c = 0; c < 6; ++c) dst[c] = f2bf(src[c]);
  dst[6] = 0; dst[7] = 0;
}

// Pack weights to bf16 [N=384][KTOT]; input-conv K range uses CPAD channel
// stride (CREAL real channels, 9 taps), zero elsewhere.
__global__ void pack_w(const float* __restrict__ Wx, const float* __restrict__ Wh,
                       unsigned short* __restrict__ wout,
                       int CPAD, int CREAL, int KXPAD, int KTOT)
{
  const int idx = blockIdx.x * blockDim.x + threadIdx.x;
  if (idx >= 384 * KTOT) return;
  const int n = idx / KTOT;
  const int k = idx - n * KTOT;
  float v = 0.f;
  if (k < KXPAD) {
    const int pos = k / CPAD;
    const int c   = k - pos * CPAD;
    if (pos < 9 && c < CREAL) v = Wx[(pos * CREAL + c) * 384 + n];
  } else {
    v = Wh[(k - KXPAD) * 384 + n];
  }
  wout[idx] = f2bf(v);
}

// ---------------------------------------------------------------------------
extern "C" void kernel_launch(void* const* d_in, const int* in_sizes, int n_in,
                              void* d_out, int out_size, void* d_ws, size_t ws_size,
                              hipStream_t stream)
{
  const float* x   = (const float*)d_in[0];
  const float* Wx0 = (const float*)d_in[1];
  const float* Wh0 = (const float*)d_in[2];
  const float* b0  = (const float*)d_in[3];
  const float* g0  = (const float*)d_in[4];
  const float* be0 = (const float*)d_in[5];
  const float* m0  = (const float*)d_in[6];
  const float* v0  = (const float*)d_in[7];
  const float* Wx1 = (const float*)d_in[8];
  const float* Wh1 = (const float*)d_in[9];
  const float* b1  = (const float*)d_in[10];
  const float* g1  = (const float*)d_in[11];
  const float* be1 = (const float*)d_in[12];
  const float* m1  = (const float*)d_in[13];
  const float* v1  = (const float*)d_in[14];
  const float* Wd  = (const float*)d_in[15];
  const float* bd  = (const float*)d_in[16];
  float* out = (float*)d_out;
  (void)in_sizes; (void)n_in; (void)out_size; (void)ws_size;

  // workspace carve-out (256B aligned); total ~21 MB — fully L2 resident
  char* ws = (char*)d_ws;
  size_t off = 0;
  auto take = [&](size_t bytes) -> char* {
    off = (off + 255) & ~(size_t)255;
    char* p = ws + off; off += bytes; return p;
  };
  unsigned short* wcat0   = (unsigned short*)take((size_t)384 * 960  * 2);
  unsigned short* wcat1   = (unsigned short*)take((size_t)384 * 1728 * 2);
  float* scale0 = (float*)take(96 * 4);
  float* shift0 = (float*)take(96 * 4);
  float* scale1 = (float*)take(96 * 4);
  float* shift1 = (float*)take(96 * 4);
  unsigned short* xpad    = (unsigned short*)take((size_t)16 * 64 * 324 * 8 * 2);
  unsigned short* h0pad   = (unsigned short*)take((size_t)16 * 324 * 96 * 2);
  unsigned short* h0bnpad = (unsigned short*)take((size_t)16 * 324 * 96 * 2);
  unsigned short* h1pad   = (unsigned short*)take((size_t)16 * 324 * 96 * 2);
  unsigned short* h1bn    = (unsigned short*)take((size_t)393216 * 2);
  float*          c0      = (float*)take((size_t)393216 * 4);
  float*          c1      = (float*)take((size_t)393216 * 4);
  float*          gbuf    = (float*)take((size_t)1572864 * 4);

  // zero everything we use (states, halos, pads) in one pass
  off = (off + 255) & ~(size_t)255;
  const int nz = (int)(off / 16);
  zero_ws<<<(nz + 255) / 256, 256, 0, stream>>>((uint4*)ws, nz);

  prep_params<<<1, 256, 0, stream>>>(g0, be0, m0, v0, g1, be1, m1, v1, bd,
                                     scale0, shift0, scale1, shift1, out);
  pack_w<<<(384 * 960  + 255) / 256, 256, 0, stream>>>(Wx0, Wh0, wcat0, 8, 6, 96, 960);
  pack_w<<<(384 * 1728 + 255) / 256, 256, 0, stream>>>(Wx1, Wh1, wcat1, 96, 96, 864, 1728);
  pad_x<<<(262144 + 255) / 256, 256, 0, stream>>>(x, xpad);

  const dim3 ggrid(64, 3);   // 4096/64 M-tiles x 384/128 N-tiles
  for (int t = 0; t < 64; ++t) {
    // layer 0 step: padded x_t (C padded to 8 -> KXPAD 96) + recurrent h0
    lstm_gate_gemm<8, 96, 960><<<ggrid, 256, 0, stream>>>(
        xpad + (size_t)t * (324 * 8), 64 * 324 * 8, h0pad, wcat0, b0, gbuf);
    lstm_pointwise<true><<<1536, 256, 0, stream>>>(gbuf, c0, h0pad, h0bnpad, scale0, shift0);
    // layer 1 step: BN0(h0_t) padded + recurrent h1
    lstm_gate_gemm<96, 864, 1728><<<ggrid, 256, 0, stream>>>(
        h0bnpad, 324 * 96, h1pad, wcat1, b1, gbuf);
    lstm_pointwise<false><<<1536, 256, 0, stream>>>(gbuf, c1, h1pad, h1bn, scale1, shift1);
    // fold BN1(h1_t) into the dense output on the fly
    dense_acc<<<16, 256, 0, stream>>>(h1bn, Wd, t, out);
  }
}
